// ResidualVectorQuantizer_30210799960621
// MI455X (gfx1250) — matrix operand order repaired and checked by reading, not compiled, not run on hardware
//
#include <hip/hip_runtime.h>
#include <hip/hip_bf16.h>

// Residual VQ for MI455X (gfx1250): bf16 WMMA distance GEMM fused with argmin,
// exact-f32 residual update / loss, deterministic reductions (no float atomics).
// Hot loop: 32x32 register tile per wave -> 4 WMMA per 8 b128 loads (2 loads/WMMA).

typedef __attribute__((ext_vector_type(16))) __bf16 v16bf;
typedef __attribute__((ext_vector_type(8)))  float  v8f;

#define RVQ_N 32768
#define RVQ_D 512
#define RVQ_Q 4
#define RVQ_K 2048
#define TILE_M 128
#define LDS_STRIDE 520   // halves: 512 + 8 pad -> row stride 1040B = 4-bank skew

__device__ __forceinline__ unsigned int bf16rne(float f) {
    unsigned int u = __float_as_uint(f);
    u += 0x7FFFu + ((u >> 16) & 1u);   // round-to-nearest-even truncation
    return u >> 16;
}

// ---- codebook f32 -> bf16 (once per launch) --------------------------------
__global__ void rvq_cvt_cb(const float* __restrict__ cb, unsigned short* __restrict__ cbh) {
    size_t i = (size_t)blockIdx.x * blockDim.x + threadIdx.x;   // float4 index
    float4 v = ((const float4*)cb)[i];
    uint2 o;
    o.x = bf16rne(v.x) | (bf16rne(v.y) << 16);
    o.y = bf16rne(v.z) | (bf16rne(v.w) << 16);
    ((uint2*)cbh)[i] = o;
}

// ---- ||e_k||^2 in f32, one wave per codebook row ---------------------------
__global__ void rvq_cnorm(const float* __restrict__ cb, float* __restrict__ cnorms) {
    int row  = (int)((blockIdx.x * blockDim.x + threadIdx.x) >> 5);   // Q*K rows
    int lane = threadIdx.x & 31;
    const float* p = cb + (size_t)row * RVQ_D;
    float s = 0.f;
    for (int i = lane * 4; i < RVQ_D; i += 32 * 4) {
        float4 v = *(const float4*)(p + i);
        s += v.x * v.x + v.y * v.y + v.z * v.z + v.w * v.w;
    }
    #pragma unroll
    for (int m = 16; m >= 1; m >>= 1) s += __shfl_xor(s, m, 32);
    if (lane == 0) cnorms[row] = s;
}

// ---- fused distance-GEMM + argmin (4 waves, 32 rows x 32 codes per wave) ---
__launch_bounds__(128)
__global__ void rvq_argmin(const float* __restrict__ r_in,
                           const unsigned short* __restrict__ cbh,   // [Q*K*D] bf16
                           const float* __restrict__ cnorms,         // [Q*K]
                           int* __restrict__ idx_out,
                           float* __restrict__ idx_out_f,            // [N*Q] float indices
                           int q) {
    __shared__ unsigned short ldsA[TILE_M * LDS_STRIDE];   // 130 KB (320 KB WGP LDS)

    const int rowbase = blockIdx.x * TILE_M;

    // Stage 128 rows x 512 f32 -> bf16 LDS (coalesced float4 reads)
    {
        const float4* src = (const float4*)(r_in + (size_t)rowbase * RVQ_D);
        for (int i = threadIdx.x; i < TILE_M * (RVQ_D / 4); i += 128) {
            int row = i >> 7;        // D/4 = 128 float4 per row
            int c4  = i & 127;
            float4 v = src[i];
            uint2 o;
            o.x = bf16rne(v.x) | (bf16rne(v.y) << 16);
            o.y = bf16rne(v.z) | (bf16rne(v.w) << 16);
            *(uint2*)&ldsA[row * LDS_STRIDE + c4 * 4] = o;
        }
    }
    __syncthreads();

    const int w    = threadIdx.x >> 5;   // 4 waves, 32 rows each
    const int lane = threadIdx.x & 31;
    const int lg   = lane >> 4;          // half-wave group: selects K sub-chunk
    const int li   = lane & 15;

    const unsigned short* ldsRow0 = &ldsA[(w * 32 + li) * LDS_STRIDE];
    const unsigned short* ldsRow1 = ldsRow0 + 16 * LDS_STRIDE;
    const size_t cb_q = (size_t)q * RVQ_K * RVQ_D;
    const float* cn_q = cnorms + (size_t)q * RVQ_K;

    float bd[2][8];
    int   bi[2][8];
    #pragma unroll
    for (int f = 0; f < 2; ++f)
        #pragma unroll
        for (int j = 0; j < 8; ++j) { bd[f][j] = 3.0e38f; bi[f][j] = 0; }

    for (int c0 = 0; c0 < RVQ_K; c0 += 32) {
        v8f acc00 = {};   // rows 0-15  x codes c0..c0+15
        v8f acc01 = {};   // rows 0-15  x codes c0+16..c0+31
        v8f acc10 = {};   // rows 16-31 x codes c0..c0+15
        v8f acc11 = {};   // rows 16-31 x codes c0+16..c0+31
        const unsigned short* b0row = cbh + cb_q + (size_t)(c0 + li) * RVQ_D;
        const unsigned short* b1row = b0row + (size_t)16 * RVQ_D;

        #pragma unroll 2
        for (int kb = 0; kb < RVQ_D; kb += 32) {
            v16bf a0, a1, b0, b1;
            // A frag (16-bit 16x32 layout): chunks at K = kb+lg*8 and K = kb+16+lg*8
            {
                const int o0 = kb + lg * 8;
                ((uint4*)&a0)[0] = *(const uint4*)(ldsRow0 + o0);
                ((uint4*)&a0)[1] = *(const uint4*)(ldsRow0 + o0 + 16);
                ((uint4*)&a1)[0] = *(const uint4*)(ldsRow1 + o0);
                ((uint4*)&a1)[1] = *(const uint4*)(ldsRow1 + o0 + 16);
            }
            // B frag (16-bit 32x16 layout): chunks at K = kb+lg*8 and K = kb+16+lg*8, column = li
            {
                const int ob = kb + lg * 8;
                ((uint4*)&b0)[0] = *(const uint4*)(b0row + ob);
                ((uint4*)&b0)[1] = *(const uint4*)(b0row + ob + 16);
                ((uint4*)&b1)[0] = *(const uint4*)(b1row + ob);
                ((uint4*)&b1)[1] = *(const uint4*)(b1row + ob + 16);
            }
            acc00 = __builtin_amdgcn_wmma_f32_16x16x32_bf16(false, a0, false, b0,
                                                            (short)0, acc00, false, false);
            acc01 = __builtin_amdgcn_wmma_f32_16x16x32_bf16(false, a0, false, b1,
                                                            (short)0, acc01, false, false);
            acc10 = __builtin_amdgcn_wmma_f32_16x16x32_bf16(false, a1, false, b0,
                                                            (short)0, acc10, false, false);
            acc11 = __builtin_amdgcn_wmma_f32_16x16x32_bf16(false, a1, false, b1,
                                                            (short)0, acc11, false, false);
        }

        const int   c0i = c0 + li;
        const int   c1i = c0i + 16;
        const float cn0 = cn_q[c0i];
        const float cn1 = cn_q[c1i];
        #pragma unroll
        for (int j = 0; j < 8; ++j) {
            // score = ||e||^2 - 2 r.e  (||r||^2 is row-constant: irrelevant to argmin)
            float s;
            s = fmaf(-2.0f, acc00[j], cn0);
            if (s < bd[0][j]) { bd[0][j] = s; bi[0][j] = c0i; }
            s = fmaf(-2.0f, acc01[j], cn1);
            if (s < bd[0][j]) { bd[0][j] = s; bi[0][j] = c1i; }
            s = fmaf(-2.0f, acc10[j], cn0);
            if (s < bd[1][j]) { bd[1][j] = s; bi[1][j] = c0i; }
            s = fmaf(-2.0f, acc11[j], cn1);
            if (s < bd[1][j]) { bd[1][j] = s; bi[1][j] = c1i; }
        }
    }

    // argmin across the 16 lanes of each half-wave (C layout: lanes0-15 M=j, 16-31 M=j+8)
    #pragma unroll
    for (int f = 0; f < 2; ++f) {
        #pragma unroll
        for (int j = 0; j < 8; ++j) {
            #pragma unroll
            for (int m = 1; m <= 8; m <<= 1) {
                float od = __shfl_xor(bd[f][j], m, 32);
                int   oi = __shfl_xor(bi[f][j], m, 32);
                if (od < bd[f][j] || (od == bd[f][j] && oi < bi[f][j])) {
                    bd[f][j] = od; bi[f][j] = oi;
                }
            }
            if (li == 0) {
                int row = rowbase + w * 32 + f * 16 + j + lg * 8;
                idx_out[row] = bi[f][j];
                idx_out_f[(size_t)row * RVQ_Q + q] = (float)bi[f][j];
            }
        }
    }
}

// ---- exact f32 residual update + per-row loss partial ----------------------
__launch_bounds__(128)
__global__ void rvq_update(const float* __restrict__ r_in,
                           const float* __restrict__ cb,       // f32 [Q,K,D]
                           const int* __restrict__ idx,
                           float* __restrict__ r_out,
                           float* __restrict__ partials,       // [Q*N]
                           int q) {
    const int n = blockIdx.x;
    const int t = threadIdx.x;
    const int best = idx[n];
    float4 rv = ((const float4*)(r_in + (size_t)n * RVQ_D))[t];
    float4 ev = ((const float4*)(cb + ((size_t)q * RVQ_K + best) * RVQ_D))[t];
    float4 rn = { rv.x - ev.x, rv.y - ev.y, rv.z - ev.z, rv.w - ev.w };
    ((float4*)(r_out + (size_t)n * RVQ_D))[t] = rn;

    float s = rn.x * rn.x + rn.y * rn.y + rn.z * rn.z + rn.w * rn.w;
    #pragma unroll
    for (int m = 16; m >= 1; m >>= 1) s += __shfl_xor(s, m, 32);
    __shared__ float ws4[4];
    if ((t & 31) == 0) ws4[t >> 5] = s;
    __syncthreads();
    if (t == 0)
        partials[(size_t)q * RVQ_N + n] = (ws4[0] + ws4[1]) + (ws4[2] + ws4[3]);
}

// ---- deterministic loss finalize -------------------------------------------
__global__ void rvq_loss(const float* __restrict__ partials, float* __restrict__ out_loss) {
    __shared__ float sh[256];
    float s = 0.f;
    for (int i = threadIdx.x; i < RVQ_Q * RVQ_N; i += 256) s += partials[i];
    sh[threadIdx.x] = s;
    __syncthreads();
    for (int step = 128; step >= 1; step >>= 1) {
        if ((int)threadIdx.x < step) sh[threadIdx.x] += sh[threadIdx.x + step];
        __syncthreads();
    }
    if (threadIdx.x == 0)
        out_loss[0] = sh[0] * (1.25f / ((float)RVQ_Q * (float)RVQ_N * (float)RVQ_D));
}

// ---- x_q = x - r_final (no need to accumulate x_q per stage) ---------------
__global__ void rvq_xq(const float* __restrict__ x, const float* __restrict__ r,
                       float* __restrict__ out) {
    size_t i = (size_t)blockIdx.x * blockDim.x + threadIdx.x;
    float4 xv = ((const float4*)x)[i];
    float4 rv = ((const float4*)r)[i];
    float4 o = { xv.x - rv.x, xv.y - rv.y, xv.z - rv.z, xv.w - rv.w };
    ((float4*)out)[i] = o;
}

extern "C" void kernel_launch(void* const* d_in, const int* in_sizes, int n_in,
                              void* d_out, int out_size, void* d_ws, size_t ws_size,
                              hipStream_t stream) {
    (void)in_sizes; (void)n_in; (void)out_size; (void)ws_size;
    const float* x  = (const float*)d_in[0];   // [N,D]
    const float* cb = (const float*)d_in[1];   // [Q,K,D]

    float* out      = (float*)d_out;
    float* out_loss = out + (size_t)RVQ_N * RVQ_D;
    float* out_idx  = out_loss + 1;            // [N,Q] as float

    char* ws = (char*)d_ws;
    float*          r        = (float*)ws;                                        // 64 MB
    unsigned short* cbh      = (unsigned short*)(ws + (size_t)RVQ_N * RVQ_D * 4); // 8 MB
    float*          cnorms   = (float*)((char*)cbh + (size_t)RVQ_Q * RVQ_K * RVQ_D * 2);
    int*            idx      = (int*)((char*)cnorms + (size_t)RVQ_Q * RVQ_K * 4);
    float*          partials = (float*)((char*)idx + (size_t)RVQ_N * 4);

    rvq_cvt_cb<<<(RVQ_Q * RVQ_K * RVQ_D / 4) / 256, 256, 0, stream>>>(cb, cbh);
    rvq_cnorm<<<(RVQ_Q * RVQ_K) / 8, 256, 0, stream>>>(cb, cnorms);

    for (int q = 0; q < RVQ_Q; ++q) {
        const float* rin = (q == 0) ? x : r;
        rvq_argmin<<<RVQ_N / TILE_M, 128, 0, stream>>>(rin, cbh, cnorms, idx, out_idx, q);
        rvq_update<<<RVQ_N, 128, 0, stream>>>(rin, cb, idx, r, partials, q);
    }

    rvq_loss<<<1, 256, 0, stream>>>(partials, out_loss);
    rvq_xq<<<(RVQ_N * RVQ_D / 4) / 256, 256, 0, stream>>>(x, r, out);
}
